// Attention_module_actor_38938173505559
// MI455X (gfx1250) — matrix operand order repaired and checked by reading, not compiled, not run on hardware
//
#include <hip/hip_runtime.h>
#include <hip/hip_bf16.h>
#include <cmath>

#define B_TOTAL 32768
#define NN 14
#define FIN 10
#define HH 3
#define DD 32
#define HD 96          // H*D
#define EPSF 1e-5f

typedef __attribute__((ext_vector_type(16))) _Float16 v16h;
typedef __attribute__((ext_vector_type(8)))  float    v8f;

// ---------------------------------------------------------------- utilities
__device__ __forceinline__ float wave_sum(float v) {
#pragma unroll
    for (int m = 16; m >= 1; m >>= 1) v += __shfl_xor(v, m, 32);
    return v;
}
__device__ __forceinline__ float eluf(float x) { return x > 0.f ? x : expm1f(x); }

// ================================================================ kernel 1:
// attention front-end. One wave32 per batch element, 2 waves per block.
__global__ __launch_bounds__(64) void frontend_kernel(
    const float* __restrict__ x,
    const float* __restrict__ k_w, const float* __restrict__ k_b,
    const float* __restrict__ q_w, const float* __restrict__ q_b,
    const float* __restrict__ v_w, const float* __restrict__ v_b,
    const float* __restrict__ kn_g, const float* __restrict__ kn_b,
    const float* __restrict__ qn_g, const float* __restrict__ qn_b,
    const float* __restrict__ vn_g, const float* __restrict__ vn_b,
    const float* __restrict__ klin_w, const float* __restrict__ klin_b,
    const float* __restrict__ qlin_w, const float* __restrict__ qlin_b,
    const float* __restrict__ alin_w, const float* __restrict__ alin_b,
    const float* __restrict__ lin1_w, const float* __restrict__ lin1_b,
    float* __restrict__ pooled)
{
    __shared__ float xs  [2][NN*FIN];
    __shared__ float kqv [2][3*HH*NN*DD];   // [tensor][h][n][d], 3*1344 floats / wave
    __shared__ float att0[2][NN*NN];
    __shared__ float att1[2][NN*NN];
    __shared__ float Esc [2][NN*HD];
    __shared__ float el  [2][NN*DD];

    const int wave = threadIdx.x >> 5;
    const int lane = threadIdx.x & 31;
    const int b = blockIdx.x * 2 + wave;

    float* XS  = xs[wave];
    float* KQV = kqv[wave];
    float* AT0 = att0[wave];
    float* AT1 = att1[wave];
    float* E   = Esc[wave];
    float* EL  = el[wave];

    for (int j = lane; j < NN*FIN; j += 32) XS[j] = x[(size_t)b * NN * FIN + j];
    __syncthreads();

    // --- QKV projections (10 -> 96) + LN partial stats; store as [h][n][d]
    float s1[3] = {0.f,0.f,0.f}, s2[3] = {0.f,0.f,0.f};
    for (int j = lane; j < NN*HD; j += 32) {      // 1344 = 42*32, uniform counts
        int n = j / HD, o = j % HD;
        float sk = k_b[o], sq = q_b[o], sv = v_b[o];
#pragma unroll
        for (int f = 0; f < FIN; f++) {
            float xv = XS[n*FIN + f];
            sk += xv * k_w[f*HD + o];
            sq += xv * q_w[f*HD + o];
            sv += xv * v_w[f*HD + o];
        }
        int st = (o >> 5) * (NN*DD) + n*DD + (o & 31);   // h*448 + n*32 + d
        KQV[0*1344 + st] = sk; KQV[1*1344 + st] = sq; KQV[2*1344 + st] = sv;
        s1[0]+=sk; s2[0]+=sk*sk; s1[1]+=sq; s2[1]+=sq*sq; s1[2]+=sv; s2[2]+=sv*sv;
    }
    __syncthreads();

    float mean[3], rs[3];
#pragma unroll
    for (int tI = 0; tI < 3; tI++) {
        float su = wave_sum(s1[tI]), sq = wave_sum(s2[tI]);
        mean[tI] = su / 1344.f;
        rs[tI] = rsqrtf(sq / 1344.f - mean[tI]*mean[tI] + EPSF);
    }
    const float* G[3]  = {kn_g, qn_g, vn_g};
    const float* Bb[3] = {kn_b, qn_b, vn_b};
    for (int j = lane; j < 1344; j += 32) {
#pragma unroll
        for (int tI = 0; tI < 3; tI++) {
            float v = (KQV[tI*1344 + j] - mean[tI]) * rs[tI];
            KQV[tI*1344 + j] = v * G[tI][j] + Bb[tI][j];   // g/b are [H,N,D] flat
        }
    }
    __syncthreads();

    // --- per-head additive attention
    for (int h = 0; h < HH; h++) {
        const float* Kh = KQV + 0*1344 + h*(NN*DD);
        const float* Qh = KQV + 1*1344 + h*(NN*DD);
        const float* Vh = KQV + 2*1344 + h*(NN*DD);

        for (int idx = lane; idx < NN*NN; idx += 32) {
            int n = idx / NN, jj = idx % NN;
            float acc = qlin_b[jj] + klin_b[jj];
            for (int d = 0; d < DD; d++)
                acc += Qh[n*DD + d]*qlin_w[d*NN + jj] + Kh[n*DD + d]*klin_w[d*NN + jj];
            AT0[idx] = eluf(acc);
        }
        __syncthreads();
        for (int idx = lane; idx < NN*NN; idx += 32) {
            int n = idx / NN, jj = idx % NN;
            float acc = alin_b[jj];
#pragma unroll
            for (int c = 0; c < NN; c++) acc += AT0[n*NN + c] * alin_w[c*NN + jj];
            AT1[idx] = acc;
        }
        __syncthreads();
        if (lane < NN) {                       // row softmax
            int n = lane;
            float mx = -3.4e38f;
#pragma unroll
            for (int c = 0; c < NN; c++) mx = fmaxf(mx, AT1[n*NN + c]);
            float e[NN], sum = 0.f;
#pragma unroll
            for (int c = 0; c < NN; c++) { e[c] = expf(AT1[n*NN + c] - mx); sum += e[c]; }
            float inv = 1.f / sum;
#pragma unroll
            for (int c = 0; c < NN; c++) AT1[n*NN + c] = e[c] * inv;
        }
        __syncthreads();
        for (int idx = lane; idx < NN*DD; idx += 32) {   // E_h = A @ V_h
            int f = idx / DD, dd = idx % DD;
            float acc = 0.f;
#pragma unroll
            for (int c = 0; c < NN; c++) acc += AT1[f*NN + c] * Vh[c*DD + dd];
            E[f*HD + h*DD + dd] = acc;                   // transpose back to [n][h*D+d]
        }
        __syncthreads();
    }

    // --- lin1 + relu + LN(no affine) + max-pool over nodes
    float t1 = 0.f, t2 = 0.f;
    for (int idx = lane; idx < NN*DD; idx += 32) {       // 448 = 14*32, uniform
        int n = idx / DD, dd = idx % DD;
        float acc = lin1_b[dd];
        for (int k2 = 0; k2 < HD; k2++) acc += E[n*HD + k2] * lin1_w[k2*DD + dd];
        acc = fmaxf(acc, 0.f);
        EL[idx] = acc;
        t1 += acc; t2 += acc * acc;
    }
    __syncthreads();
    float su = wave_sum(t1), sq = wave_sum(t2);
    float m2  = su / 448.f;
    float rsv = rsqrtf(sq / 448.f - m2*m2 + EPSF);
    {
        int d = lane;                                    // DD == 32 lanes
        float mx = -3.4e38f;
#pragma unroll
        for (int n = 0; n < NN; n++) mx = fmaxf(mx, (EL[n*DD + d] - m2) * rsv);
        pooled[(size_t)b * DD + d] = mx;
    }
}

// ================================================================ kernel 2:
// WMMA MLP tail. One wave = 16 batch rows; 4 waves/block.
union AFrag { unsigned int u[8]; v16h h; };

// A fragment from LDS activations [16][stride] f16.
// 16-bit A 16x32 layout (contiguous, matches B): lane<16 -> K 0..15; lane>=16 -> K 16..31.
__device__ __forceinline__ v16h load_a_frag(const _Float16* act, int lane, int kBase, int stride) {
    int row = lane & 15;
    int hi  = lane >> 4;
    AFrag f;
    const unsigned int* u32 = reinterpret_cast<const unsigned int*>(act);
#pragma unroll
    for (int i = 0; i < 8; i++) {
        int K0 = 2*i + 16*hi;
        f.u[i] = u32[(row * stride + kBase + K0) >> 1];
    }
    return f.h;
}

// B fragment from pre-transposed f16 weights WT[Npad][Kpad]:
// lane<16 -> col N=lane, K=kBase..kBase+15; lane>=16 -> K += 16. One v16h load.
__device__ __forceinline__ v16h load_b_frag(const _Float16* WT, int lane, int nt, int kBase, int Kpad) {
    int n  = nt*16 + (lane & 15);
    int hi = lane >> 4;
    return *reinterpret_cast<const v16h*>(WT + (size_t)n * Kpad + kBase + 16*hi);
}

// act: 0 = elu, 1 = tanh. dst stride fixed 128.
__device__ __forceinline__ void mlp_layer_lds(
    const _Float16* src, _Float16* dst, const _Float16* WT, const float* bias,
    int lane, int kChunks, int nTiles, int Kpad, int Nvalid, int actKind)
{
    int col = lane & 15;
    int hi  = lane >> 4;
    for (int nt = 0; nt < nTiles; nt++) {
        v8f c = {};
        for (int kc = 0; kc < kChunks; kc++) {
            v16h a = load_a_frag(src, lane, kc*32, 128);
            v16h w = load_b_frag(WT, lane, nt, kc*32, Kpad);
            c = __builtin_amdgcn_wmma_f32_16x16x32_f16(false, a, false, w, (short)0, c, false, false);
        }
        int n = nt*16 + col;
        float bv = (n < Nvalid) ? bias[n] : 0.f;
#pragma unroll
        for (int r = 0; r < 8; r++) {
            float v = c[r] + bv;
            v = (actKind == 0) ? (v > 0.f ? v : expm1f(v)) : tanhf(v);
            dst[(r + 8*hi) * 128 + n] = (_Float16)v;     // act(0)=0 keeps padding clean
        }
    }
}

__device__ __forceinline__ void mlp_layer_out(
    const _Float16* src, const _Float16* WT, const float* bias,
    int lane, int kChunks, int Kpad, int Nvalid,
    float* outp, int outStride, int rowBase,
    const unsigned char* mask /* non-null => tanh+mask, null => sigmoid */)
{
    int col = lane & 15;
    int hi  = lane >> 4;
    v8f c = {};
    for (int kc = 0; kc < kChunks; kc++) {
        v16h a = load_a_frag(src, lane, kc*32, 128);
        v16h w = load_b_frag(WT, lane, 0, kc*32, Kpad);
        c = __builtin_amdgcn_wmma_f32_16x16x32_f16(false, a, false, w, (short)0, c, false, false);
    }
    if (col < Nvalid) {
        float bv = bias[col];
#pragma unroll
        for (int r = 0; r < 8; r++) {
            int gRow = rowBase + r + 8*hi;
            float v = c[r] + bv;
            if (mask) {
                v = tanhf(v);
                v = mask[(size_t)gRow * outStride + col] ? v : -1e8f;
            } else {
                v = 1.f / (1.f + expf(-v));
            }
            outp[(size_t)gRow * outStride + col] = v;
        }
    }
}

__global__ __launch_bounds__(128) void mlp_kernel(
    const float* __restrict__ pooled,
    const _Float16* __restrict__ lin2T, const float* __restrict__ lin2_b,
    const _Float16* __restrict__ l4T, const float* __restrict__ l4_b,
    const _Float16* __restrict__ l5T, const float* __restrict__ l5_b,
    const _Float16* __restrict__ l6T, const float* __restrict__ l6_b,
    const _Float16* __restrict__ l7T, const float* __restrict__ l7_b,
    const _Float16* __restrict__ l8T, const float* __restrict__ l8_b,
    const _Float16* __restrict__ l9T, const float* __restrict__ l9_b,
    const unsigned char* __restrict__ mask,
    float* __restrict__ out1, float* __restrict__ out2)
{
    __shared__ _Float16 actA[4][16*128];
    __shared__ _Float16 actY[4][16*128];
    __shared__ _Float16 actB[4][16*128];

    const int wave = threadIdx.x >> 5;
    const int lane = threadIdx.x & 31;
    const int rowBase = blockIdx.x * 64 + wave * 16;

    _Float16* aA = actA[wave];
    _Float16* aY = actY[wave];
    _Float16* aB = actB[wave];

    for (int j = lane; j < 16*128; j += 32) {
        aA[j] = (_Float16)0.f; aY[j] = (_Float16)0.f; aB[j] = (_Float16)0.f;
    }
    __syncthreads();

    for (int j = lane; j < 16*32; j += 32) {             // stage pooled feats (K=32)
        int r = j >> 5, k = j & 31;
        aA[r*128 + k] = (_Float16)pooled[(size_t)(rowBase + r) * 32 + k];
    }
    __syncthreads();

    // y = elu(feat @ lin2 + b)           M=16 K=32  N=112(100)
    mlp_layer_lds(aA, aY, lin2T, lin2_b, lane, 1, 7, 32, 100, 0);
    __syncthreads();
    // head 1: tanh -> tanh -> tanh+mask
    mlp_layer_lds(aY, aA, l4T, l4_b, lane, 4, 7, 128, 100, 1);
    __syncthreads();
    mlp_layer_lds(aA, aB, l5T, l5_b, lane, 4, 7, 128, 100, 1);
    __syncthreads();
    mlp_layer_out(aB, l6T, l6_b, lane, 4, 128, 9, out1, 9, rowBase, mask);
    __syncthreads();
    // head 2: tanh -> tanh -> sigmoid
    mlp_layer_lds(aY, aA, l7T, l7_b, lane, 4, 7, 128, 100, 1);
    __syncthreads();
    mlp_layer_lds(aA, aB, l8T, l8_b, lane, 4, 7, 128, 100, 1);
    __syncthreads();
    mlp_layer_out(aB, l9T, l9_b, lane, 4, 128, 8, out2, 8, rowBase, nullptr);
}

// ================================================================ kernel 0:
// transpose+pad weights f32 [K][N] -> f16 [Npad][Kpad] (zero padded).
__global__ void prep_wT_kernel(const float* __restrict__ src, _Float16* __restrict__ dst,
                               int K, int Nv, int Kpad, int Npad)
{
    int idx = blockIdx.x * blockDim.x + threadIdx.x;
    if (idx >= Kpad * Npad) return;
    int n = idx / Kpad, k = idx - n * Kpad;
    float v = (k < K && n < Nv) ? src[k * Nv + n] : 0.f;
    dst[idx] = (_Float16)v;
}

// ================================================================ launch
extern "C" void kernel_launch(void* const* d_in, const int* in_sizes, int n_in,
                              void* d_out, int out_size, void* d_ws, size_t ws_size,
                              hipStream_t stream)
{
    const float* x      = (const float*)d_in[0];
    const unsigned char* mask = (const unsigned char*)d_in[1];   // jax bool = 1 byte
    const float* k_w    = (const float*)d_in[2];  const float* k_b    = (const float*)d_in[3];
    const float* q_w    = (const float*)d_in[4];  const float* q_b    = (const float*)d_in[5];
    const float* v_w    = (const float*)d_in[6];  const float* v_b    = (const float*)d_in[7];
    const float* kn_g   = (const float*)d_in[8];  const float* kn_b   = (const float*)d_in[9];
    const float* qn_g   = (const float*)d_in[10]; const float* qn_b   = (const float*)d_in[11];
    const float* vn_g   = (const float*)d_in[12]; const float* vn_b   = (const float*)d_in[13];
    const float* klin_w = (const float*)d_in[14]; const float* klin_b = (const float*)d_in[15];
    const float* qlin_w = (const float*)d_in[16]; const float* qlin_b = (const float*)d_in[17];
    const float* alin_w = (const float*)d_in[18]; const float* alin_b = (const float*)d_in[19];
    const float* lin1_w = (const float*)d_in[20]; const float* lin1_b = (const float*)d_in[21];
    const float* lin2_w = (const float*)d_in[22]; const float* lin2_b = (const float*)d_in[23];
    const float* l4_w   = (const float*)d_in[24]; const float* l4_b   = (const float*)d_in[25];
    const float* l5_w   = (const float*)d_in[26]; const float* l5_b   = (const float*)d_in[27];
    const float* l6_w   = (const float*)d_in[28]; const float* l6_b   = (const float*)d_in[29];
    const float* l7_w   = (const float*)d_in[30]; const float* l7_b   = (const float*)d_in[31];
    const float* l8_w   = (const float*)d_in[32]; const float* l8_b   = (const float*)d_in[33];
    const float* l9_w   = (const float*)d_in[34]; const float* l9_b   = (const float*)d_in[35];

    // workspace layout (≈4.33 MB total)
    char* ws = (char*)d_ws;
    float* pooled = (float*)ws;                              size_t off = (size_t)B_TOTAL * 32 * 4;
    _Float16* lin2T = (_Float16*)(ws + off);                 off += (size_t)112 * 32  * 2;
    _Float16* l4T   = (_Float16*)(ws + off);                 off += (size_t)112 * 128 * 2;
    _Float16* l5T   = (_Float16*)(ws + off);                 off += (size_t)112 * 128 * 2;
    _Float16* l6T   = (_Float16*)(ws + off);                 off += (size_t)16  * 128 * 2;
    _Float16* l7T   = (_Float16*)(ws + off);                 off += (size_t)112 * 128 * 2;
    _Float16* l8T   = (_Float16*)(ws + off);                 off += (size_t)112 * 128 * 2;
    _Float16* l9T   = (_Float16*)(ws + off);                 off += (size_t)16  * 128 * 2;

    auto prep = [&](const float* src, _Float16* dst, int K, int Nv, int Kpad, int Npad) {
        int total = Kpad * Npad;
        prep_wT_kernel<<<(total + 255) / 256, 256, 0, stream>>>(src, dst, K, Nv, Kpad, Npad);
    };
    prep(lin2_w, lin2T, 32,  100, 32,  112);
    prep(l4_w,   l4T,   100, 100, 128, 112);
    prep(l5_w,   l5T,   100, 100, 128, 112);
    prep(l6_w,   l6T,   100, 9,   128, 16);
    prep(l7_w,   l7T,   100, 100, 128, 112);
    prep(l8_w,   l8T,   100, 100, 128, 112);
    prep(l9_w,   l9T,   100, 8,   128, 16);

    frontend_kernel<<<B_TOTAL / 2, 64, 0, stream>>>(
        x, k_w, k_b, q_w, q_b, v_w, v_b,
        kn_g, kn_b, qn_g, qn_b, vn_g, vn_b,
        klin_w, klin_b, qlin_w, qlin_b, alin_w, alin_b,
        lin1_w, lin1_b, pooled);

    float* out1 = (float*)d_out;
    float* out2 = out1 + (size_t)B_TOTAL * 9;
    mlp_kernel<<<B_TOTAL / 64, 128, 0, stream>>>(
        pooled,
        lin2T, lin2_b, l4T, l4_b, l5T, l5_b, l6T, l6_b,
        l7T, l7_b, l8T, l8_b, l9T, l9_b,
        mask, out1, out2);
}
